// Encoder_43748536877120
// MI455X (gfx1250) — matrix operand order repaired and checked by reading, compiled, not run on hardware
//
#include <hip/hip_runtime.h>
#include <hip/hip_bf16.h>
#include <math.h>

// ---------------- dims ----------------
#define BB   32
#define TT   512
#define CDIM 512            // E == C == 512
#define HH   256
#define G4H  1024           // 4*H
#define KTAP 5
#define TP   520            // T + 4 pad each side
#define NTOT (BB*TT)        // 16384
#define RS   80             // LDS slab row stride in halfs (160B, 32B-aligned, bank-spread)

typedef __attribute__((ext_vector_type(16))) _Float16 v16h;
typedef __attribute__((ext_vector_type(8)))  _Float16 v8h;
typedef __attribute__((ext_vector_type(8)))  float    v8f;
typedef __attribute__((ext_vector_type(4)))  int      v4i;

#ifndef __has_builtin
#define __has_builtin(x) 0
#endif

#if __has_builtin(__builtin_amdgcn_global_load_async_to_lds_b128)
#define HAVE_ASYNC 1
#else
#define HAVE_ASYNC 0
#endif

#if HAVE_ASYNC
// signature (from clang diagnostic): param0 = v4i in global AS; dst in LDS AS
typedef __attribute__((address_space(1))) v4i GV4;
typedef __attribute__((address_space(3))) v4i LV4;
__device__ __forceinline__ void async_cp16(const _Float16* g, _Float16* l) {
  __builtin_amdgcn_global_load_async_to_lds_b128((GV4*)g, (LV4*)l, 0, 0);
}
template <int N>
__device__ __forceinline__ void wait_asynccnt() {
#if __has_builtin(__builtin_amdgcn_s_wait_asynccnt)
  __builtin_amdgcn_s_wait_asynccnt(N);
#else
  asm volatile("s_wait_asynccnt %0" ::"n"(N) : "memory");
#endif
}
#endif

// copy 16B global -> LDS (async when available; sync fallback keeps same structure)
__device__ __forceinline__ void cp16(const _Float16* g, _Float16* l) {
#if HAVE_ASYNC
  async_cp16(g, l);
#else
  *(v8h*)l = *(const v8h*)g;
#endif
}

// A-fragment (16x32 f16, MxK): lane l holds m = l&15.
// e=0..15 -> K = (l>>4)*16 + e
__device__ __forceinline__ v16h load_afrag(const _Float16* p) {
  v8h lo = *(const v8h*)(p);
  v8h hi = *(const v8h*)(p + 8);
  return __builtin_shufflevector(lo, hi, 0,1,2,3,4,5,6,7,8,9,10,11,12,13,14,15);
}
// B-fragment (32x16 f16, KxN): lane l holds n = l&15, elements e -> K = (l>>4)*16 + e
// -> one contiguous 32B load per lane from an [n][k] row-major tile.

__device__ __forceinline__ v8f wmma_f16(v16h a, v16h b, v8f c) {
  return __builtin_amdgcn_wmma_f32_16x16x32_f16(false, a, false, b, (short)0, c, false, false);
}

// ---------------- prep kernels ----------------
__global__ void k_zero_pads(_Float16* xa, _Float16* xb) {
  int i = blockIdx.x * blockDim.x + threadIdx.x;
  if (i >= BB * 8 * CDIM) return;
  int ci = i & (CDIM - 1);
  int r  = (i >> 9) & 7;
  int b  = i >> 12;
  int row = (r < 4) ? r : (TP - 8 + r);   // rows 0..3 and 516..519
  size_t idx = ((size_t)b * TP + row) * CDIM + ci;
  xa[idx] = (_Float16)0.f;
  xb[idx] = (_Float16)0.f;
}

__global__ void k_embed(const int* __restrict__ seq, const float* __restrict__ emb,
                        _Float16* __restrict__ xpad) {
  size_t i = (size_t)blockIdx.x * blockDim.x + threadIdx.x;
  if (i >= (size_t)BB * TT * CDIM) return;
  int ci = (int)(i & (CDIM - 1));
  int t  = (int)((i >> 9) & (TT - 1));
  int b  = (int)(i >> 18);
  int v  = seq[b * TT + t];
  xpad[((size_t)b * TP + t + 4) * CDIM + ci] = (_Float16)emb[(size_t)v * CDIM + ci];
}

// conv w [Cout][Cin][K] f32 -> [K][Cout][Cin] f16
__global__ void k_cvt_convw(const float* __restrict__ w, _Float16* __restrict__ wt) {
  int i = blockIdx.x * blockDim.x + threadIdx.x;
  if (i >= CDIM * CDIM * KTAP) return;
  int k  = i % KTAP;
  int ci = (i / KTAP) & (CDIM - 1);
  int co = i / (KTAP * CDIM);
  wt[((size_t)k * CDIM + co) * CDIM + ci] = (_Float16)w[i];
}

__global__ void k_cvt_f16(const float* __restrict__ a, _Float16* __restrict__ o, int n) {
  int i = blockIdx.x * blockDim.x + threadIdx.x;
  if (i < n) o[i] = (_Float16)a[i];
}

__global__ void k_prep_bn(const float* __restrict__ gamma, const float* __restrict__ beta,
                          const float* __restrict__ mean,  const float* __restrict__ var,
                          float* __restrict__ scale, float* __restrict__ shift) {
  int c = blockIdx.x * blockDim.x + threadIdx.x;
  if (c >= CDIM) return;
  float inv = rsqrtf(var[c] + 1e-5f);
  float s = inv * gamma[c];
  scale[c] = s;
  shift[c] = beta[c] - mean[c] * s;
}

__global__ void k_bias_sum(const float* __restrict__ b1, const float* __restrict__ b2,
                           float* __restrict__ o, int n) {
  int i = blockIdx.x * blockDim.x + threadIdx.x;
  if (i < n) o[i] = b1[i] + b2[i];
}

// ---------------- dilated conv as LDS-staged GEMM + bias + BN + ReLU ----------------
// xin/xout: [B][TP][C] f16 padded; wtap: [5][C][C] f16
// block 256 = 8 waves: M=128 (16 co/wave), N=64 (4 n-tiles shared by all waves).
// Per K-step (32 ci): stage a 72-row x 32-ci slab (serves all 5 dilated taps) into
// LDS with async b128 copies, double-buffered; B fragments then read from LDS.
__global__ __launch_bounds__(256)
void k_conv(const _Float16* __restrict__ xin, _Float16* __restrict__ xout,
            const _Float16* __restrict__ wtap, const float* __restrict__ cb,
            const float* __restrict__ scale, const float* __restrict__ shift) {
  __shared__ _Float16 slab[2][72 * RS];     // 2 x 11520B
  const int tid  = threadIdx.x;
  const int lane = tid & 31;
  const int wave = tid >> 5;
  const int hi   = lane >> 4;
  const int ln   = lane & 15;
  const int m0   = blockIdx.y * 128 + wave * 16;     // output-channel tile base
  const int n_base = blockIdx.x * 64;                // n = b*T + t, 64 consecutive t, one b
  const int b      = n_base >> 9;
  const int t_base = n_base & (TT - 1);
  const _Float16* gslab = xin + ((size_t)b * TP + t_base) * CDIM;  // padded row t_base

  // two 16B transfers per thread per stage (288 needed; extras clamped -> benign dup)
  const int r1 = tid >> 2, q1 = tid & 3;
  int r2 = (tid + 256) >> 2; if (r2 > 71) r2 = 71;
  const int q2 = tid & 3;

  v8f acc[4] = {};
  const _Float16* wrow0 = wtap + (size_t)(m0 + ln) * CDIM;   // lane's co row, tap 0

#define CONV_STAGE(buf, kb)                                                     \
  do {                                                                          \
    cp16(gslab + r1 * CDIM + (kb) * 32 + q1 * 8, &slab[buf][r1 * RS + q1 * 8]); \
    cp16(gslab + r2 * CDIM + (kb) * 32 + q2 * 8, &slab[buf][r2 * RS + q2 * 8]); \
  } while (0)

  CONV_STAGE(0, 0);
  for (int kb = 0; kb < CDIM / 32; ++kb) {
    const int cur = kb & 1;
    if (kb + 1 < CDIM / 32) {
      CONV_STAGE(cur ^ 1, kb + 1);
#if HAVE_ASYNC
      wait_asynccnt<2>();        // current slab done; next stage stays in flight
#endif
    } else {
#if HAVE_ASYNC
      wait_asynccnt<0>();
#endif
    }
    __syncthreads();
#pragma unroll
    for (int tap = 0; tap < KTAP; ++tap) {
      v16h a = load_afrag(wrow0 + (size_t)tap * CDIM * CDIM + kb * 32 + hi * 16);
#pragma unroll
      for (int ni = 0; ni < 4; ++ni) {
        // slab row for (j = ni*16+ln, tap): j + 4 + 2*(tap-2) = j + 2*tap
        const _Float16* bp = &slab[cur][(ni * 16 + ln + 2 * tap) * RS + hi * 16];
        v16h bf = *(const v16h*)bp;
        acc[ni] = wmma_f16(a, bf, acc[ni]);
      }
    }
    __syncthreads();   // protect slab[cur] before it is re-staged
  }
#undef CONV_STAGE

#pragma unroll
  for (int ni = 0; ni < 4; ++ni) {
    int t = t_base + ni * 16 + ln;
    _Float16* orow = xout + ((size_t)b * TP + t + 4) * CDIM;
#pragma unroll
    for (int r = 0; r < 8; ++r) {
      int co = m0 + r + 8 * hi;
      float y = (acc[ni][r] + cb[co]) * scale[co] + shift[co];
      orow[co] = (_Float16)(y > 0.f ? y : 0.f);
    }
  }
}

// ---------------- LSTM input GEMM: gp[(t*B+b)*4H + g] = x@Wih^T + bih + bhh ----------------
// Same LDS-staged scheme, 64 rows/slab (one 16B transfer per thread per stage).
__global__ __launch_bounds__(256)
void k_gates_pre(const _Float16* __restrict__ xin, const _Float16* __restrict__ Wih,
                 const float* __restrict__ bsum, float* __restrict__ gp) {
  __shared__ _Float16 slab[2][64 * RS];     // 2 x 10240B
  const int tid  = threadIdx.x;
  const int lane = tid & 31;
  const int wave = tid >> 5;
  const int hi   = lane >> 4;
  const int ln   = lane & 15;
  const int m0   = blockIdx.y * 128 + wave * 16;   // gate-row tile base (M = 1024)
  const int n_base = blockIdx.x * 64;              // n = t*B + b

  const int rr = tid >> 2, qq = tid & 3;           // slab row / 16B quarter
  const int n_r = n_base + rr;
  const int b_r = n_r & (BB - 1), t_r = n_r >> 5;
  const _Float16* grow = xin + ((size_t)b_r * TP + t_r + 4) * CDIM;

  v8f acc[4] = {};
  const _Float16* wrow = Wih + (size_t)(m0 + ln) * CDIM;

  cp16(grow + qq * 8, &slab[0][rr * RS + qq * 8]);
  for (int kb = 0; kb < CDIM / 32; ++kb) {
    const int cur = kb & 1;
    if (kb + 1 < CDIM / 32) {
      cp16(grow + (kb + 1) * 32 + qq * 8, &slab[cur ^ 1][rr * RS + qq * 8]);
#if HAVE_ASYNC
      wait_asynccnt<1>();
#endif
    } else {
#if HAVE_ASYNC
      wait_asynccnt<0>();
#endif
    }
    __syncthreads();
    v16h a = load_afrag(wrow + kb * 32 + hi * 16);
#pragma unroll
    for (int ni = 0; ni < 4; ++ni) {
      const _Float16* bp = &slab[cur][(ni * 16 + ln) * RS + hi * 16];
      v16h bf = *(const v16h*)bp;
      acc[ni] = wmma_f16(a, bf, acc[ni]);
    }
    __syncthreads();
  }
#pragma unroll
  for (int ni = 0; ni < 4; ++ni) {
    int n = n_base + ni * 16 + ln;
    float* orow = gp + (size_t)n * G4H;
#pragma unroll
    for (int r = 0; r < 8; ++r) {
      int g = m0 + r + 8 * hi;
      orow[g] = acc[ni][r] + bsum[g];
    }
  }
}

// ---------------- persistent bidirectional zoneout-LSTM recurrence ----------------
// one block per direction; 1024 threads = 32 waves; h kept as f16 in LDS for WMMA B frags
#define LSTM_SMEM (BB*G4H*4 + BB*HH*4*2 + BB*HH*2)   // gates + h + c + hf16 = 212992 B
__global__ __launch_bounds__(1024)
void k_lstm(const float* __restrict__ gp, const _Float16* __restrict__ Whh,
            float* __restrict__ out) {
  extern __shared__ unsigned char smem_raw[];
  float*    gates = (float*)smem_raw;                  // [32][1024]
  float*    hs    = gates + BB * G4H;                  // [32][256]
  float*    cs    = hs + BB * HH;                      // [32][256]
  _Float16* hf    = (_Float16*)(cs + BB * HH);         // [32][256]

  const int dir  = blockIdx.x;
  const float*    gpd  = gp  + (size_t)dir * TT * BB * G4H;
  const _Float16* Whhd = Whh + (size_t)dir * G4H * HH;

  const int tid  = threadIdx.x;
  const int lane = tid & 31, wave = tid >> 5;
  const int hi = lane >> 4, ln = lane & 15;
  const int m0a = wave * 32;                           // this wave's 2 gate-row tiles

  for (int i = tid; i < BB * HH; i += 1024) {
    hs[i] = 0.f; cs[i] = 0.f; hf[i] = (_Float16)0.f;
  }
  __syncthreads();

  for (int step = 0; step < TT; ++step) {
    const int t = dir ? (TT - 1 - step) : step;

    // gates = gp[t] + h @ Whh^T   (M=1024, N=32, K=256)
    v8f acc[2][2];
#pragma unroll
    for (int mi = 0; mi < 2; ++mi)
#pragma unroll
      for (int ni = 0; ni < 2; ++ni) {
        int b = ni * 16 + ln;
        const float* p = gpd + ((size_t)(t * BB + b)) * G4H + m0a + mi * 16 + 8 * hi;
        v8f a0;
#pragma unroll
        for (int r = 0; r < 8; ++r) a0[r] = p[r];
        acc[mi][ni] = a0;
      }
#pragma unroll
    for (int kb = 0; kb < HH / 32; ++kb) {
      v16h af[2];
#pragma unroll
      for (int mi = 0; mi < 2; ++mi)
        af[mi] = load_afrag(Whhd + (size_t)(m0a + mi * 16 + ln) * HH + kb * 32 + hi * 16);
#pragma unroll
      for (int ni = 0; ni < 2; ++ni) {
        v16h bf = *(const v16h*)(hf + (ni * 16 + ln) * HH + kb * 32 + hi * 16);
#pragma unroll
        for (int mi = 0; mi < 2; ++mi)
          acc[mi][ni] = wmma_f16(af[mi], bf, acc[mi][ni]);
      }
    }
#pragma unroll
    for (int mi = 0; mi < 2; ++mi)
#pragma unroll
      for (int ni = 0; ni < 2; ++ni) {
        int b = ni * 16 + ln;
#pragma unroll
        for (int r = 0; r < 8; ++r)
          gates[b * G4H + m0a + mi * 16 + r + 8 * hi] = acc[mi][ni][r];
      }
    __syncthreads();

    // elementwise cell + eval-mode zoneout
    for (int cell = tid; cell < BB * HH; cell += 1024) {
      int b = cell >> 8, h = cell & (HH - 1);
      const float* gb = gates + b * G4H;
      float ig = gb[h], fg = gb[HH + h], gg = gb[2 * HH + h], og = gb[3 * HH + h];
      float c_old = cs[cell], h_old = hs[cell];
      float si = 1.f / (1.f + __expf(-ig));
      float sf = 1.f / (1.f + __expf(-fg));
      float so = 1.f / (1.f + __expf(-og));
      float cn = sf * c_old + si * tanhf(gg);
      float hn = so * tanhf(cn);
      float cz = 0.9f * cn + 0.1f * c_old;
      float hz = 0.9f * hn + 0.1f * h_old;
      cs[cell] = cz; hs[cell] = hz; hf[cell] = (_Float16)hz;
      out[((size_t)b * TT + t) * (2 * HH) + dir * HH + h] = hz;
    }
    __syncthreads();
  }
}

// ---------------- launch ----------------
extern "C" void kernel_launch(void* const* d_in, const int* in_sizes, int n_in,
                              void* d_out, int out_size, void* d_ws, size_t ws_size,
                              hipStream_t stream) {
  (void)in_sizes; (void)n_in; (void)out_size; (void)ws_size;
  const int*   seq   = (const int*)  d_in[0];
  const float* emb   = (const float*)d_in[1];
  const float* w1    = (const float*)d_in[2];
  const float* b1    = (const float*)d_in[3];
  const float* w2    = (const float*)d_in[4];
  const float* b2    = (const float*)d_in[5];
  const float* bng   = (const float*)d_in[6];
  const float* bnb   = (const float*)d_in[7];
  const float* bnm   = (const float*)d_in[8];
  const float* bnv   = (const float*)d_in[9];
  const float* fWih  = (const float*)d_in[10];
  const float* fWhh  = (const float*)d_in[11];
  const float* fbih  = (const float*)d_in[12];
  const float* fbhh  = (const float*)d_in[13];
  const float* bWih  = (const float*)d_in[14];
  const float* bWhh  = (const float*)d_in[15];
  const float* bbih  = (const float*)d_in[16];
  const float* bbhh  = (const float*)d_in[17];
  float* out = (float*)d_out;

  // -------- workspace carve-up --------
  char* ws = (char*)d_ws;
  size_t off = 0;
  auto take = [&](size_t bytes) { char* p = ws + off; off += (bytes + 255) & ~(size_t)255; return p; };
  _Float16* xpadA = (_Float16*)take((size_t)BB * TP * CDIM * 2);
  _Float16* xpadB = (_Float16*)take((size_t)BB * TP * CDIM * 2);
  _Float16* w1h   = (_Float16*)take((size_t)KTAP * CDIM * CDIM * 2);
  _Float16* w2h   = (_Float16*)take((size_t)KTAP * CDIM * CDIM * 2);
  _Float16* WihF  = (_Float16*)take((size_t)G4H * CDIM * 2);
  _Float16* WihB  = (_Float16*)take((size_t)G4H * CDIM * 2);
  _Float16* WhhFB = (_Float16*)take((size_t)2 * G4H * HH * 2);  // fwd then bwd, contiguous
  float*    bsumF = (float*)take(G4H * 4);
  float*    bsumB = (float*)take(G4H * 4);
  float*    bnsc  = (float*)take(CDIM * 4);
  float*    bnsh  = (float*)take(CDIM * 4);
  float*    gpFB  = (float*)take((size_t)2 * TT * BB * G4H * 4); // fwd then bwd, contiguous

  // -------- prep --------
  k_zero_pads<<<(BB * 8 * CDIM + 255) / 256, 256, 0, stream>>>(xpadA, xpadB);
  k_embed<<<(int)(((size_t)BB * TT * CDIM + 255) / 256), 256, 0, stream>>>(seq, emb, xpadA);
  k_prep_bn<<<(CDIM + 255) / 256, 256, 0, stream>>>(bng, bnb, bnm, bnv, bnsc, bnsh);
  k_cvt_convw<<<(CDIM * CDIM * KTAP + 255) / 256, 256, 0, stream>>>(w1, w1h);
  k_cvt_convw<<<(CDIM * CDIM * KTAP + 255) / 256, 256, 0, stream>>>(w2, w2h);
  k_cvt_f16<<<(G4H * CDIM + 255) / 256, 256, 0, stream>>>(fWih, WihF, G4H * CDIM);
  k_cvt_f16<<<(G4H * CDIM + 255) / 256, 256, 0, stream>>>(bWih, WihB, G4H * CDIM);
  k_cvt_f16<<<(G4H * HH + 255) / 256, 256, 0, stream>>>(fWhh, WhhFB, G4H * HH);
  k_cvt_f16<<<(G4H * HH + 255) / 256, 256, 0, stream>>>(bWhh, WhhFB + (size_t)G4H * HH, G4H * HH);
  k_bias_sum<<<(G4H + 255) / 256, 256, 0, stream>>>(fbih, fbhh, bsumF, G4H);
  k_bias_sum<<<(G4H + 255) / 256, 256, 0, stream>>>(bbih, bbhh, bsumB, G4H);

  // -------- conv stack (conv1, conv2, conv2 again; shared BN) --------
  dim3 cgrid(NTOT / 64, CDIM / 128);
  k_conv<<<cgrid, 256, 0, stream>>>(xpadA, xpadB, w1h, b1, bnsc, bnsh);
  k_conv<<<cgrid, 256, 0, stream>>>(xpadB, xpadA, w2h, b2, bnsc, bnsh);
  k_conv<<<cgrid, 256, 0, stream>>>(xpadA, xpadB, w2h, b2, bnsc, bnsh);

  // -------- LSTM input GEMMs (both directions) --------
  dim3 ggrid(NTOT / 64, G4H / 128);
  k_gates_pre<<<ggrid, 256, 0, stream>>>(xpadB, WihF, bsumF, gpFB);
  k_gates_pre<<<ggrid, 256, 0, stream>>>(xpadB, WihB, bsumB, gpFB + (size_t)TT * BB * G4H);

  // -------- recurrent bidirectional LSTM (persistent, 1 block/dir) --------
  k_lstm<<<2, 1024, LSTM_SMEM, stream>>>(gpFB, WhhFB, out);
}
